// AttentionAugmentedConv2d_2534030704753
// MI455X (gfx1250) — compile-verified, matrix operand order checked by reading, not run on hardware
//
#include <hip/hip_runtime.h>

typedef __attribute__((ext_vector_type(16))) __bf16 v16bf;
typedef __attribute__((ext_vector_type(8)))  __bf16 v8bf;
typedef __attribute__((ext_vector_type(8)))  float  v8f;
typedef __attribute__((ext_vector_type(4)))  unsigned int v4u;
typedef __attribute__((ext_vector_type(8)))  int v8i;
typedef __attribute__((ext_vector_type(4)))  int v4i;

#define DEVINL __device__ __forceinline__

#if defined(__has_builtin)
#if __has_builtin(__builtin_amdgcn_tensor_load_to_lds)
#define HAVE_TDM 1
#endif
#endif
#ifndef HAVE_TDM
#define HAVE_TDM 0
#endif

// ---------------------------------------------------------------- helpers ---
DEVINL unsigned short f2bf_bits(float f) {
  unsigned int u = __builtin_bit_cast(unsigned int, f);
  unsigned int r = u + 0x7FFFu + ((u >> 16) & 1u);   // round-to-nearest-even
  return (unsigned short)(r >> 16);
}
DEVINL __bf16 f2bf(float f) { return __builtin_bit_cast(__bf16, f2bf_bits(f)); }

// Load a 16x32 bf16 fragment (A: rows=M, or B via N-major "Bt": rows=N).
// Per CDNA5 ISA 16-bit A layout: lanes 0-15 hold K {0..7, 16..23},
// lanes 16-31 hold K {8..15, 24..31}; row = lane % 16.
DEVINL v16bf load_frag(const __bf16* base, int ld) {
  const int lane = threadIdx.x & 31;
  const int row  = lane & 15;
  const int ko   = (lane >> 4) << 3;           // 0 or 8
  const __bf16* p = base + row * ld;
  union { v16bf v; v8bf h[2]; } u;
  u.h[0] = *(const v8bf*)(p + ko);
  u.h[1] = *(const v8bf*)(p + 16 + ko);
  return u.v;
}

DEVINL v8f wmma_bf16(v16bf a, v16bf b, v8f c) {
  return __builtin_amdgcn_wmma_f32_16x16x32_bf16(false, a, false, b,
                                                 (short)0, c, false, false);
}

#if HAVE_TDM
// Issue one TDM 2D tile load: 32 bf16 (64B) x `rows` rows, row stride
// `stride_elems` bf16 in global, into LDS with 16B padding appended per 64B
// row (-> 40-half padded LDS rows, matching load_frag's ld=40).
DEVINL void tdm_load_tile_bf16(unsigned lds_byte_addr, const __bf16* gsrc,
                               unsigned rows, unsigned stride_elems) {
  unsigned long long ga = (unsigned long long)(size_t)gsrc;
  v4u g0;
  g0[0] = 1u;                                              // count=1 (valid)
  g0[1] = lds_byte_addr;                                   // lds_addr
  g0[2] = (unsigned)ga;                                    // global_addr[31:0]
  g0[3] = (unsigned)((ga >> 32) & 0x01FFFFFFull) | (2u << 30); // addr hi | type=2
  const unsigned td0 = 0x7FFFFFFFu, td1 = 0x7FFFFFFFu;     // huge dims: no OOB
  v8i g1;
  g1[0] = (int)((1u << 16)      // data_size = 2 bytes
              | (1u << 20)      // pad_enable
              | (3u << 22)      // pad_interval: 16 DWORDs (=64B row)
              | (3u << 25));    // pad_amount:   4 DWORDs (=16B pad)
  g1[1] = (int)((td0 & 0xFFFFu) << 16);                    // tensor_dim0 lo
  g1[2] = (int)((td0 >> 16) | ((td1 & 0xFFFFu) << 16));    // dim0 hi | dim1 lo
  g1[3] = (int)((td1 >> 16) | (32u << 16));                // dim1 hi | tile_dim0=32
  g1[4] = (int)rows;                                       // tile_dim1
  g1[5] = (int)stride_elems;                               // dim0_stride[31:0]
  g1[6] = 0;                                               // stride hi, dim1_stride
  g1[7] = 0;
  v4i gz4 = {0, 0, 0, 0};
  v8i gz8 = {0, 0, 0, 0, 0, 0, 0, 0};
  __builtin_amdgcn_tensor_load_to_lds(g0, g1, gz4, gz4, gz8, 0);
}
#endif

// ------------------------------------------------------------ pack kernels ---
__global__ void pack_w1_kernel(const float* __restrict__ w_qkv,
                               const float* __restrict__ w_conv,
                               const float* __restrict__ b_qkv,
                               const float* __restrict__ b_conv,
                               __bf16* __restrict__ Wpack,
                               float* __restrict__ bias1) {
  size_t idx = (size_t)blockIdx.x * blockDim.x + threadIdx.x;
  const size_t total = (size_t)1024 * 2304;
  if (idx < total) {
    size_t oc = idx / 2304;
    float v = (oc < 768) ? w_qkv[idx] : w_conv[idx - (size_t)768 * 2304];
    Wpack[idx] = f2bf(v);
  }
  if (idx < 1024)
    bias1[idx] = (idx < 768) ? b_qkv[idx] : b_conv[idx - 768];
}

__global__ void pack_watt_kernel(const float* __restrict__ w_att,
                                 __bf16* __restrict__ Watt) {
  int idx = blockIdx.x * blockDim.x + threadIdx.x;
  if (idx < 256 * 256) Watt[idx] = f2bf(w_att[idx]);
}

// im2col (N-major): Xcol[n][k], n = b*1024 + y*32 + x, k = ci*9 + ky*3 + kx
__global__ void im2col_kernel(const float* __restrict__ x,
                              __bf16* __restrict__ Xcol) {
  int n  = blockIdx.x;                 // 0..8191
  int b  = n >> 10, hw = n & 1023;
  int y  = hw >> 5, xx = hw & 31;
  __bf16* row = Xcol + (size_t)n * 2304;
  for (int k = threadIdx.x; k < 2304; k += blockDim.x) {
    int ci = k / 9, rem = k - ci * 9;
    int ky = rem / 3, kx = rem - ky * 3;
    int iy = y + ky - 1, ix = xx + kx - 1;
    float v = 0.f;
    if ((unsigned)iy < 32u && (unsigned)ix < 32u)
      v = x[(size_t)(b * 256 + ci) * 1024 + iy * 32 + ix];
    row[k] = f2bf(v);
  }
}

// --------------------------------------------------------- generic WMMA GEMM
// C[M][N] = A[M][K](bf16,row-major) * Bt[N][K](bf16,N-major) + bias[m]
// mode 0: C[m*N + n]
// mode 1: n=(b,hw): C[((b*512 + chan_off + m) << 10) + hw]   (final output)
// LDS staging is double-buffered via the Tensor Data Mover: wave 0 issues a
// D# pair (A tile + B tile) for step k+1, waits TENSORcnt<=2 so step k's pair
// has landed, then the workgroup barrier releases the compute waves.
__global__ __launch_bounds__(256)
void gemm_bf16_wmma(const __bf16* __restrict__ A,
                    const __bf16* __restrict__ Bt,
                    const float*  __restrict__ bias,
                    float*        __restrict__ C,
                    int M, int N, int K, int mode, int chan_off) {
  __shared__ __bf16 As[2][128][40];
  __shared__ __bf16 Bs[2][128][40];

  const int tid  = threadIdx.x;
  const int lane = tid & 31;
  const int w    = tid >> 5;           // 0..7
  const int wm   = w & 3;              // 4 waves along M (32 rows each)
  const int wn   = w >> 2;             // 2 waves along N (64 cols each)
  const int m0   = blockIdx.y * 128;
  const int n0   = blockIdx.x * 128;
  const int nk   = K >> 5;

  v8f zero = {};
  v8f acc[2][4];
#pragma unroll
  for (int i = 0; i < 2; ++i)
#pragma unroll
    for (int j = 0; j < 4; ++j) acc[i][j] = zero;

#if HAVE_TDM
  const unsigned bufBytes = 128u * 40u * 2u;               // 10240
  const unsigned ldsA = (unsigned)(size_t)&As[0][0][0];
  const unsigned ldsB = (unsigned)(size_t)&Bs[0][0][0];
  if (w == 0) {
    tdm_load_tile_bf16(ldsA, A  + (size_t)m0 * K, 128u, (unsigned)K);
    tdm_load_tile_bf16(ldsB, Bt + (size_t)n0 * K, 128u, (unsigned)K);
  }
#endif

  for (int kt = 0; kt < nk; ++kt) {
    const int cur = kt & 1;
#if HAVE_TDM
    if (w == 0) {
      if (kt + 1 < nk) {
        const unsigned nb = (unsigned)(cur ^ 1) * bufBytes;
        tdm_load_tile_bf16(ldsA + nb, A  + (size_t)m0 * K + (kt + 1) * 32,
                           128u, (unsigned)K);
        tdm_load_tile_bf16(ldsB + nb, Bt + (size_t)n0 * K + (kt + 1) * 32,
                           128u, (unsigned)K);
        __builtin_amdgcn_s_wait_tensorcnt(2);  // pair kt done; kt+1 in flight
      } else {
        __builtin_amdgcn_s_wait_tensorcnt(0);
      }
    }
    __syncthreads();
#else
    __syncthreads();
    const int k0 = kt * 32;
#pragma unroll
    for (int it = 0; it < 2; ++it) {
      int c  = tid + it * 256;
      int m  = c >> 2;
      int kc = (c & 3) * 8;
      *(uint4*)&As[cur][m][kc] =
          *(const uint4*)(A  + (size_t)(m0 + m) * K + k0 + kc);
      *(uint4*)&Bs[cur][m][kc] =
          *(const uint4*)(Bt + (size_t)(n0 + m) * K + k0 + kc);
    }
    __syncthreads();
#endif

    v16bf af[2], bfr[4];
#pragma unroll
    for (int i = 0; i < 2; ++i)
      af[i]  = load_frag(&As[cur][wm * 32 + i * 16][0], 40);
#pragma unroll
    for (int j = 0; j < 4; ++j)
      bfr[j] = load_frag(&Bs[cur][wn * 64 + j * 16][0], 40);
#pragma unroll
    for (int i = 0; i < 2; ++i)
#pragma unroll
      for (int j = 0; j < 4; ++j)
        acc[i][j] = wmma_bf16(af[i], bfr[j], acc[i][j]);
    __syncthreads();
  }

  const int lo = lane & 15, hi = lane >> 4;
#pragma unroll
  for (int i = 0; i < 2; ++i)
#pragma unroll
    for (int j = 0; j < 4; ++j)
#pragma unroll
      for (int r = 0; r < 8; ++r) {
        int m = m0 + wm * 32 + i * 16 + r + 8 * hi;
        int n = n0 + wn * 64 + j * 16 + lo;
        float v = acc[i][j][r] + bias[m];
        if (mode == 0) {
          C[(size_t)m * N + n] = v;
        } else {
          int b = n >> 10, hw = n & 1023;
          C[(((size_t)b * 512 + chan_off + m) << 10) + hw] = v;
        }
      }
}

// ------------------------------------------------------ flash attention ----
// Per (b,h), 64 query rows per block.  conv1 rows: q=[0,256) k=[256,512)
// v=[512,768), each row = [b*1024 + hw].  Output scattered to AttBt[n][c]
// (N-major operand of the 1x1 projection) with the reference's reshape
// mapping: c = h*32 + xq, n = b*1024 + yq*32 + d.
__global__ __launch_bounds__(128)
void flash_attn_kernel(const float* __restrict__ conv1,
                       const float* __restrict__ key_rel_w,
                       const float* __restrict__ key_rel_h,
                       __bf16* __restrict__ AttBt) {
  __shared__ __bf16 Qs[64][40];        // [m][d], scaled
  __shared__ __bf16 Ks[128][40];       // [n][d]  (Bt for S)
  __shared__ __bf16 Vs[32][136];       // [d][n]  (Bt for O)
  __shared__ __bf16 Ps[64][136];       // softmaxed probs (A for O)
  __shared__ __bf16 RelW[64][64];      // q . key_rel_w[j]
  __shared__ __bf16 RelH[64][64];

  const int tid  = threadIdx.x;
  const int lane = tid & 31;
  const int w    = tid >> 5;           // 0..3, 16 query rows each
  const int lo   = lane & 15, hi = lane >> 4;
  const int m0   = blockIdx.x * 64;
  const int bh   = blockIdx.y;
  const int b    = bh >> 3, h = bh & 7;
  const size_t colbase = (size_t)b * 1024;
  const float qscale = 0.17677669529663687f;   // 1/sqrt(32)

  // ---- load Q tile (scaled) ----
  for (int i = tid; i < 64 * 32; i += 128) {
    int d = i >> 6, m = i & 63;
    Qs[m][d] =
        f2bf(conv1[(size_t)(h * 32 + d) * 8192 + colbase + m0 + m] * qscale);
  }
  __syncthreads();

  // ---- relative-position dot tables (q already scaled) ----
  for (int s = tid; s < 2 * 64 * 64; s += 128) {
    int which = s >> 12;               // 0 -> RelW, 1 -> RelH
    int m = (s >> 6) & 63;
    int j = s & 63;
    float acc = 0.f;
    if (j < 63) {
      const float* tab = which ? key_rel_h : key_rel_w;
#pragma unroll
      for (int d = 0; d < 32; ++d)
        acc += (float)Qs[m][d] * tab[j * 32 + d];
    }
    if (which) RelH[m][j] = f2bf(acc);
    else       RelW[m][j] = f2bf(acc);
  }
  __syncthreads();

  v16bf qa = load_frag(&Qs[w * 16][0], 40);

  v8f zero = {};
  v8f o[2] = {zero, zero};
  float mrow[8], lrow[8];
#pragma unroll
  for (int r = 0; r < 8; ++r) { mrow[r] = -3.4e38f; lrow[r] = 0.f; }

  for (int nt = 0; nt < 8; ++nt) {
    const int n0 = nt * 128;
    __syncthreads();                    // protect Ks/Vs from prev iteration
    for (int i = tid; i < 128 * 32; i += 128) {
      int d = i >> 7, n = i & 127;
      Ks[n][d] =
          f2bf(conv1[(size_t)(256 + h * 32 + d) * 8192 + colbase + n0 + n]);
      Vs[d][n] =
          f2bf(conv1[(size_t)(512 + h * 32 + d) * 8192 + colbase + n0 + n]);
    }
    __syncthreads();

    // ---- S = Q K^T : one WMMA per 16-key tile (K=32 = DKH) ----
    v8f st[8];
#pragma unroll
    for (int t = 0; t < 8; ++t) {
      v16bf kb = load_frag(&Ks[t * 16][0], 40);
      st[t] = wmma_bf16(qa, kb, zero);
    }

    // ---- bias + online softmax (rows r and r+8 across lane halves) ----
    float alpha[8];
#pragma unroll
    for (int r = 0; r < 8; ++r) {
      int ml = w * 16 + r + 8 * hi;
      int mq = m0 + ml, yq = mq & 31, xq = mq >> 5;
      float vmax = -3.4e38f;
#pragma unroll
      for (int t = 0; t < 8; ++t) {
        int ng = n0 + t * 16 + lo, yk = ng & 31, xk = ng >> 5;
        float sv = st[t][r] + (float)RelW[ml][yk - yq + 31] +
                   (float)RelH[ml][xk - xq + 31];
        st[t][r] = sv;
        vmax = fmaxf(vmax, sv);
      }
      vmax = fmaxf(vmax, __shfl_xor(vmax, 1, 32));
      vmax = fmaxf(vmax, __shfl_xor(vmax, 2, 32));
      vmax = fmaxf(vmax, __shfl_xor(vmax, 4, 32));
      vmax = fmaxf(vmax, __shfl_xor(vmax, 8, 32));
      float mn = fmaxf(mrow[r], vmax);
      alpha[r] = __expf(mrow[r] - mn);
      float ps = 0.f;
#pragma unroll
      for (int t = 0; t < 8; ++t) {
        float p = __expf(st[t][r] - mn);
        st[t][r] = p;
        ps += p;
      }
      ps += __shfl_xor(ps, 1, 32);
      ps += __shfl_xor(ps, 2, 32);
      ps += __shfl_xor(ps, 4, 32);
      ps += __shfl_xor(ps, 8, 32);
      lrow[r] = lrow[r] * alpha[r] + ps;
      mrow[r] = mn;
    }
#pragma unroll
    for (int dt = 0; dt < 2; ++dt)
#pragma unroll
      for (int r = 0; r < 8; ++r) o[dt][r] *= alpha[r];

    // ---- stash P (bf16) for A-fragment loads (wave-private rows) ----
#pragma unroll
    for (int t = 0; t < 8; ++t)
#pragma unroll
      for (int r = 0; r < 8; ++r)
        Ps[w * 16 + r + 8 * hi][t * 16 + lo] = f2bf(st[t][r]);

    // ---- O += P V^T : K=128 -> 4 k-steps x 2 d-tiles ----
#pragma unroll
    for (int ks = 0; ks < 4; ++ks) {
      v16bf pa = load_frag(&Ps[w * 16][ks * 32], 136);
#pragma unroll
      for (int dt = 0; dt < 2; ++dt) {
        v16bf vb = load_frag(&Vs[dt * 16][ks * 32], 136);
        o[dt] = wmma_bf16(pa, vb, o[dt]);
      }
    }
  }

  // ---- epilogue: normalize + scatter with the reference reshape mapping ----
#pragma unroll
  for (int r = 0; r < 8; ++r) lrow[r] = 1.f / lrow[r];
#pragma unroll
  for (int dt = 0; dt < 2; ++dt)
#pragma unroll
    for (int r = 0; r < 8; ++r) {
      int ml = w * 16 + r + 8 * hi;
      int mq = m0 + ml, yq = mq & 31, xq = mq >> 5;
      int dg = dt * 16 + lo;
      size_t n = colbase + yq * 32 + dg;     // pixel index for 1x1 conv
      int    c = h * 32 + xq;                // channel index
      AttBt[n * 256 + c] = f2bf(o[dt][r] * lrow[r]);
    }
}

// ------------------------------------------------------------------ launch --
extern "C" void kernel_launch(void* const* d_in, const int* in_sizes, int n_in,
                              void* d_out, int out_size, void* d_ws,
                              size_t ws_size, hipStream_t stream) {
  (void)in_sizes; (void)n_in; (void)out_size; (void)ws_size;
  const float* x      = (const float*)d_in[0];
  const float* w_conv = (const float*)d_in[1];
  const float* b_conv = (const float*)d_in[2];
  const float* w_qkv  = (const float*)d_in[3];
  const float* b_qkv  = (const float*)d_in[4];
  const float* w_att  = (const float*)d_in[5];
  const float* b_att  = (const float*)d_in[6];
  const float* krw    = (const float*)d_in[7];
  const float* krh    = (const float*)d_in[8];
  float* out = (float*)d_out;

  char* ws = (char*)d_ws;
  size_t off = 0;
  auto wsalloc = [&](size_t bytes) -> void* {
    void* p = (void*)(ws + off);
    off += (bytes + 255) & ~(size_t)255;
    return p;
  };
  __bf16* Xcol  = (__bf16*)wsalloc((size_t)8192 * 2304 * 2);  // im2col, N-major
  __bf16* Wpack = (__bf16*)wsalloc((size_t)1024 * 2304 * 2);  // qkv || conv
  float*  bias1 = (float*) wsalloc((size_t)1024 * 4);
  float*  Conv1 = (float*) wsalloc((size_t)768 * 8192 * 4);   // qkv activations
  __bf16* AttBt = (__bf16*)wsalloc((size_t)8192 * 256 * 2);   // attn out, N-major
  __bf16* Watt  = (__bf16*)wsalloc((size_t)256 * 256 * 2);

  pack_w1_kernel<<<dim3((1024 * 2304 + 255) / 256), 256, 0, stream>>>(
      w_qkv, w_conv, b_qkv, b_conv, Wpack, bias1);
  pack_watt_kernel<<<dim3(256), 256, 0, stream>>>(w_att, Watt);
  im2col_kernel<<<dim3(8192), 256, 0, stream>>>(x, Xcol);

  // qkv conv -> staging (rows 0..767)
  gemm_bf16_wmma<<<dim3(64, 6), 256, 0, stream>>>(
      Wpack, Xcol, bias1, Conv1, 768, 8192, 2304, 0, 0);
  // xo conv -> output channels [0,256)
  gemm_bf16_wmma<<<dim3(64, 2), 256, 0, stream>>>(
      Wpack + (size_t)768 * 2304, Xcol, bias1 + 768, out, 256, 8192, 2304, 1, 0);
  // fused attention (64 bh x 16 m-tiles)
  flash_attn_kernel<<<dim3(16, 64), 128, 0, stream>>>(Conv1, krw, krh, AttBt);
  // 1x1 projection -> output channels [256,512)
  gemm_bf16_wmma<<<dim3(64, 2), 256, 0, stream>>>(
      Watt, AttBt, b_att, out, 256, 8192, 256, 1, 256);
}
